// SAPModule_56831007261316
// MI455X (gfx1250) — matrix operand order repaired and checked by reading, not compiled, not run on hardware
//
#include <hip/hip_runtime.h>

// SAP self-attention (Q=K=V=feat^T), flash-attention style, bf16x3 WMMA,
// K/V tiles streamed to LDS by the CDNA5 Tensor Data Mover (double-buffered).
// B=8, C=256, N=64*64=4096. scale = 1/sqrt(256) = 0.0625.
#define BATCH   8
#define CCH     256
#define NPOS    4096
#define BM      128     // query rows per workgroup (16 per wave, 8 waves)
#define BN      32      // key columns per tile
#define THREADS 256     // 8 wave32s
#define NJT     (NPOS / BN)

#define NEG_INF (-1.0e9f)

typedef __attribute__((ext_vector_type(16))) __bf16         v16bf;
typedef __attribute__((ext_vector_type(8)))  float          v8f;
typedef __attribute__((ext_vector_type(4)))  unsigned int   v4u;
typedef __attribute__((ext_vector_type(4)))  unsigned short v4us;
typedef __attribute__((ext_vector_type(4)))  unsigned int   u32x4;
typedef __attribute__((ext_vector_type(8)))  int            i32x8;
typedef __attribute__((ext_vector_type(4)))  int            i32x4;

union FragU { v4u q[2]; v16bf v; };   // 32 bytes: one bf16 A/B WMMA operand

__device__ __forceinline__ unsigned int f2u(float f) {
  union { float f; unsigned int u; } x; x.f = f; return x.u;
}
__device__ __forceinline__ float u2f(unsigned int u) {
  union { unsigned int u; float f; } x; x.u = u; return x.f;
}
// round-to-nearest-even float -> bf16 (raw u16)
__device__ __forceinline__ unsigned short f2bf(float f) {
  unsigned int u = f2u(f);
  u = u + 0x7FFFu + ((u >> 16) & 1u);
  return (unsigned short)(u >> 16);
}
__device__ __forceinline__ float bf2f(unsigned short s) {
  return u2f(((unsigned int)s) << 16);
}

// Issue one TDM load: 2D tile [tile_dim1=CCH rows x tile_dim0=BN cols] of f32,
// row stride NPOS elements, from `gaddr` into LDS byte offset `lds_off`.
// 6-arg builtin form (clang-23 / therock-10.0 headers).
__device__ __forceinline__ void tdm_load_tile(unsigned int lds_off,
                                              unsigned long long gaddr) {
  u32x4 g0;
  g0.x = 1u;                                      // count=1, is_restore=0
  g0.y = lds_off;                                 // lds_addr (bytes)
  g0.z = (unsigned int)(gaddr & 0xFFFFFFFFull);   // global_addr[31:0]
  g0.w = (unsigned int)((gaddr >> 32) & 0x01FFFFFFull) | 0x80000000u; // addr[56:32] | type=2
  i32x8 g1;
  g1.s0 = 0x00020000;   // data_size=2 (4 bytes)
  g1.s1 = 0x10000000;   // tensor_dim0 = 4096 (low 16 in [31:16])
  g1.s2 = 0x01000000;   // tensor_dim1 = 256  (low 16 in [31:16])
  g1.s3 = 0x00200000;   // tile_dim0 = 32     (in [31:16])
  g1.s4 = 0x00000100;   // tile_dim1 = 256
  g1.s5 = 0x00001000;   // tensor_dim0_stride = 4096
  g1.s6 = 0;
  g1.s7 = 0;
  i32x4 gz4 = {0, 0, 0, 0};
  i32x8 gz8 = {0, 0, 0, 0, 0, 0, 0, 0};
  __builtin_amdgcn_tensor_load_to_lds(g0, g1, gz4, gz4, gz8, 0);
}

__global__ __launch_bounds__(THREADS, 1)
void sap_flash_attn(const float* __restrict__ x,
                    const float* __restrict__ thrp,
                    float* __restrict__ out) {
  // LDS staging (272 KB of the 320 KB CDNA5 WGP LDS)
  __shared__ unsigned short qThi[BM * CCH];        // Q^T  [m][c]  (A operand, GEMM1)
  __shared__ unsigned short qTlo[BM * CCH];
  __shared__ unsigned short kThi[BN * CCH];        // K^T  [n][c]  (B operand, GEMM1)
  __shared__ unsigned short kTlo[BN * CCH];
  __shared__ unsigned short kvhi[CCH * BN];        // V    [c][n]  (B operand, GEMM2)
  __shared__ unsigned short kvlo[CCH * BN];
  __shared__ unsigned short pThi[8 * 16 * BN];     // P per-wave   (A operand, GEMM2)
  __shared__ unsigned short pTlo[8 * 16 * BN];
  __shared__ float          kraw[2][CCH * BN];     // raw f32 K tile, TDM double buffer

  const int tid  = threadIdx.x;
  const int wid  = tid >> 5;
  const int lane = tid & 31;
  const int lh   = lane >> 4;    // lane-half (wave32 WMMA layout)
  const int l16  = lane & 15;
  const int b    = blockIdx.y;
  const int i0   = blockIdx.x * BM;
  const float thr = thrp[0];

  const float* X = x   + (size_t)b * CCH * NPOS;
  float*       Y = out + (size_t)b * CCH * NPOS;

  // ---- prologue: wave 0 kicks off the TDM for the first K tile ----
  if (wid == 0) {
    tdm_load_tile((unsigned int)(size_t)&kraw[0][0],
                  (unsigned long long)(size_t)X);
  }

  // ---- stage Q block: qT[m][c] = X[c][i0+m], split into bf16 hi+lo ----
  for (int it = 0; it < (BM * CCH) / (4 * THREADS); ++it) {   // 32 iters
    const int idx = it * THREADS + tid;
    const int c  = idx >> 5;          // 32 float4 per 128-wide row
    const int jq = (idx & 31) << 2;
    const float4 v = *reinterpret_cast<const float4*>(X + (size_t)c * NPOS + i0 + jq);
    const float e[4] = {v.x, v.y, v.z, v.w};
#pragma unroll
    for (int k = 0; k < 4; ++k) {
      const unsigned short hi = f2bf(e[k]);
      const unsigned short lo = f2bf(e[k] - bf2f(hi));
      qThi[(jq + k) * CCH + c] = hi;
      qTlo[(jq + k) * CCH + c] = lo;
    }
  }

  // ---- per-wave state: O accumulator [16 rows x 256 ch], row stats ----
  v8f accO[16];
#pragma unroll
  for (int ct = 0; ct < 16; ++ct)
#pragma unroll
    for (int r = 0; r < 8; ++r) accO[ct][r] = 0.0f;
  float mrow[8], lrow[8];
#pragma unroll
  for (int r = 0; r < 8; ++r) { mrow[r] = -3.0e38f; lrow[r] = 0.0f; }

  const unsigned short* qh = qThi + (wid * 16 + l16) * CCH + lh * 8;
  const unsigned short* ql = qTlo + (wid * 16 + l16) * CCH + lh * 8;

  int cur = 0;
  for (int jt = 0; jt < NJT; ++jt) {       // 128 key tiles
    // ---- wait for this tile's TDM, then convert raw f32 -> bf16 hi/lo ----
    if (wid == 0) __builtin_amdgcn_s_wait_tensorcnt(0);
    __syncthreads();   // kraw[cur] ready; previous iteration's kT/kv readers done

    for (int it = 0; it < (BN * CCH) / (4 * THREADS); ++it) {  // 8 iters
      const int idx = it * THREADS + tid;
      const int c  = idx >> 3;
      const int jn = (idx & 7) << 2;
      const float4 v = *reinterpret_cast<const float4*>(&kraw[cur][c * BN + jn]);
      const float e[4] = {v.x, v.y, v.z, v.w};
      unsigned short hi[4], lo[4];
#pragma unroll
      for (int k = 0; k < 4; ++k) {
        hi[k] = f2bf(e[k]);
        lo[k] = f2bf(e[k] - bf2f(hi[k]));
        kThi[(jn + k) * CCH + c] = hi[k];
        kTlo[(jn + k) * CCH + c] = lo[k];
      }
      *reinterpret_cast<v4us*>(&kvhi[c * BN + jn]) = (v4us){hi[0], hi[1], hi[2], hi[3]};
      *reinterpret_cast<v4us*>(&kvlo[c * BN + jn]) = (v4us){lo[0], lo[1], lo[2], lo[3]};
    }

    // ---- prefetch next tile into the other buffer while we compute ----
    if (wid == 0 && jt + 1 < NJT) {
      tdm_load_tile((unsigned int)(size_t)&kraw[cur ^ 1][0],
                    (unsigned long long)(size_t)(X + (size_t)(jt + 1) * BN));
    }
    __syncthreads();   // kT/kv ready

    // ---- GEMM1: S[16 x 32] = Q^T K / 16   (bf16x3, K-dim = 256) ----
    v8f accS[2];
#pragma unroll
    for (int t = 0; t < 2; ++t)
#pragma unroll
      for (int r = 0; r < 8; ++r) accS[t][r] = 0.0f;

#pragma unroll
    for (int ks = 0; ks < 8; ++ks) {
      FragU aH, aL;
      aH.q[0] = *reinterpret_cast<const v4u*>(qh + ks * 32);
      aH.q[1] = *reinterpret_cast<const v4u*>(qh + ks * 32 + 16);
      aL.q[0] = *reinterpret_cast<const v4u*>(ql + ks * 32);
      aL.q[1] = *reinterpret_cast<const v4u*>(ql + ks * 32 + 16);
#pragma unroll
      for (int t = 0; t < 2; ++t) {
        const unsigned short* kh = kThi + (t * 16 + l16) * CCH + ks * 32 + lh * 8;
        const unsigned short* kl = kTlo + (t * 16 + l16) * CCH + ks * 32 + lh * 8;
        FragU bH, bL;
        bH.q[0] = *reinterpret_cast<const v4u*>(kh);
        bH.q[1] = *reinterpret_cast<const v4u*>(kh + 16);
        bL.q[0] = *reinterpret_cast<const v4u*>(kl);
        bL.q[1] = *reinterpret_cast<const v4u*>(kl + 16);
        accS[t] = __builtin_amdgcn_wmma_f32_16x16x32_bf16(false, aH.v, false, bH.v, (short)0, accS[t], false, false);
        accS[t] = __builtin_amdgcn_wmma_f32_16x16x32_bf16(false, aH.v, false, bL.v, (short)0, accS[t], false, false);
        accS[t] = __builtin_amdgcn_wmma_f32_16x16x32_bf16(false, aL.v, false, bH.v, (short)0, accS[t], false, false);
      }
    }

    // ---- scale, threshold mask, online softmax ----
#pragma unroll
    for (int t = 0; t < 2; ++t)
#pragma unroll
      for (int r = 0; r < 8; ++r) {
        float s = accS[t][r] * 0.0625f;
        accS[t][r] = (s < thr) ? NEG_INF : s;
      }

    float fac[8];
#pragma unroll
    for (int r = 0; r < 8; ++r) {
      float v = fmaxf(accS[0][r], accS[1][r]);
      v = fmaxf(v, __shfl_xor(v, 1));
      v = fmaxf(v, __shfl_xor(v, 2));
      v = fmaxf(v, __shfl_xor(v, 4));
      v = fmaxf(v, __shfl_xor(v, 8));     // stays within 16-lane half = WMMA row group
      const float mn = fmaxf(mrow[r], v);
      const float f  = __expf(mrow[r] - mn);
      mrow[r] = mn;
      lrow[r] *= f;
      fac[r] = f;
    }
#pragma unroll
    for (int t = 0; t < 2; ++t)
#pragma unroll
      for (int r = 0; r < 8; ++r)
        accS[t][r] = __expf(accS[t][r] - mrow[r]);
#pragma unroll
    for (int r = 0; r < 8; ++r) {
      float s = accS[0][r] + accS[1][r];
      s += __shfl_xor(s, 1);
      s += __shfl_xor(s, 2);
      s += __shfl_xor(s, 4);
      s += __shfl_xor(s, 8);
      lrow[r] += s;
    }
#pragma unroll
    for (int ct = 0; ct < 16; ++ct)
#pragma unroll
      for (int r = 0; r < 8; ++r) accO[ct][r] *= fac[r];

    // ---- P: C/D layout -> A layout via per-wave private LDS (in-order, no barrier) ----
    unsigned short* ph = pThi + wid * (16 * BN);
    unsigned short* pl = pTlo + wid * (16 * BN);
#pragma unroll
    for (int t = 0; t < 2; ++t)
#pragma unroll
      for (int r = 0; r < 8; ++r) {
        const float p = accS[t][r];
        const unsigned short hi = f2bf(p);
        const unsigned short lo = f2bf(p - bf2f(hi));
        const int off = (r + lh * 8) * BN + t * 16 + l16;
        ph[off] = hi;
        pl[off] = lo;
      }

    FragU pH, pL;
    {
      const unsigned short* pa = ph + l16 * BN + lh * 8;
      const unsigned short* pb = pl + l16 * BN + lh * 8;
      pH.q[0] = *reinterpret_cast<const v4u*>(pa);
      pH.q[1] = *reinterpret_cast<const v4u*>(pa + 16);
      pL.q[0] = *reinterpret_cast<const v4u*>(pb);
      pL.q[1] = *reinterpret_cast<const v4u*>(pb + 16);
    }

    // ---- GEMM2: O[16 x 256] += P[16 x 32] * V[32 x 256]  (bf16x3) ----
#pragma unroll
    for (int ct = 0; ct < 16; ++ct) {
      const unsigned short* vh = kvhi + (ct * 16 + l16) * BN + lh * 8;
      const unsigned short* vl = kvlo + (ct * 16 + l16) * BN + lh * 8;
      FragU vH, vL;
      vH.q[0] = *reinterpret_cast<const v4u*>(vh);
      vH.q[1] = *reinterpret_cast<const v4u*>(vh + 16);
      vL.q[0] = *reinterpret_cast<const v4u*>(vl);
      vL.q[1] = *reinterpret_cast<const v4u*>(vl + 16);
      accO[ct] = __builtin_amdgcn_wmma_f32_16x16x32_bf16(false, pH.v, false, vH.v, (short)0, accO[ct], false, false);
      accO[ct] = __builtin_amdgcn_wmma_f32_16x16x32_bf16(false, pH.v, false, vL.v, (short)0, accO[ct], false, false);
      accO[ct] = __builtin_amdgcn_wmma_f32_16x16x32_bf16(false, pL.v, false, vH.v, (short)0, accO[ct], false, false);
    }

    cur ^= 1;
  }

  // ---- normalize by row sums and store ----
#pragma unroll
  for (int ct = 0; ct < 16; ++ct) {
    const int c = ct * 16 + l16;
#pragma unroll
    for (int r = 0; r < 8; ++r) {
      const int row = r + lh * 8;
      Y[(size_t)c * NPOS + i0 + wid * 16 + row] = accO[ct][r] / lrow[r];
    }
  }
}

extern "C" void kernel_launch(void* const* d_in, const int* in_sizes, int n_in,
                              void* d_out, int out_size, void* d_ws, size_t ws_size,
                              hipStream_t stream) {
  const float* x    = (const float*)d_in[0];
  const float* thr  = (const float*)d_in[1];
  float*       out  = (float*)d_out;
  (void)in_sizes; (void)n_in; (void)out_size; (void)d_ws; (void)ws_size;

  dim3 grid(NPOS / BM, BATCH);   // 32 x 8 = 256 workgroups
  sap_flash_attn<<<grid, THREADS, 0, stream>>>(x, thr, out);
}
